// FeaturePropagation_11003706212692
// MI455X (gfx1250) — matrix operand order repaired and checked by reading, not compiled, not run on hardware
//
#include <hip/hip_runtime.h>
#include <stdint.h>

// ---------------------------------------------------------------------------
// FeaturePropagation (PointNet++ FP layer) for MI455X / gfx1250 (wave32, WMMA)
//
// Pipeline:
//   1) pack_weights:    W1,W2 (f32) -> WMMA A-fragments (f16)
//   2) knn_kernel:      3-NN dists + full reciprocal-sum weights
//   3) build_xpack1:    interp(256ch) ++ skip(128ch) -> WMMA B-fragments (f16)
//   4) gemm_frag:       y1 = W1 @ X1  (v_wmma_f32_16x16x32_f16, f32 accum)
//                       + fused per-block BN partial sums (shfl_xor reduce)
//   5) stats_finalize:  BN1 mean/invstd from partials (deterministic)
//   6) bn_relu_pack:    BN+ReLU -> X2 B-fragments (f16)
//   7) gemm_frag:       y2 = W2 @ X2 (+ fused BN2 partials)
//   8) stats_finalize:  BN2 stats
//   9) bn_relu_out:     BN+ReLU -> d_out [B,Co,Ns] f32 row-major
// ---------------------------------------------------------------------------

typedef __attribute__((ext_vector_type(16))) _Float16 v16h;
typedef __attribute__((ext_vector_type(8)))  float    v8f;

#define B_   16
#define NP_  1024
#define NS_  4096
#define CP_  256
#define CS_  128
#define CO_  256
#define K1_  (CP_ + CS_)   // 384
#define KT1_ 12            // 384/32
#define KT2_ 8             // 256/32
#define NT_  (NS_/16)      // 256 N-tiles per batch
#define MT_  (CO_/16)      // 16 M-tiles
#define GBLK ((NT_/8)*B_)  // 512 GEMM blocks

// Fragment storage: 256 dwords per 16x16(x32) tile, dword index = lane*8 + v.
// A-frag (16x32 f16): M = lane%16, Kpair = kt*32 + 16*(lane/16) + 2*v
// B-frag (32x16 f16): N = lane%16, Kpair = kt*32 + 16*(lane/16) + 2*v
// D-frag (16x16 f32): M = v + 8*(lane/16), N = lane%16

union H2 { _Float16 h[2]; uint32_t u; };

static __device__ __forceinline__ uint32_t pack2(float a, float b) {
  H2 p; p.h[0] = (_Float16)a; p.h[1] = (_Float16)b; return p.u;
}

// ---------------- 1) weight packing ----------------------------------------
__global__ void __launch_bounds__(256)
fp_pack_weights(const float* __restrict__ W1, const float* __restrict__ W2,
                uint32_t* __restrict__ Wp1, uint32_t* __restrict__ Wp2) {
  int tid = blockIdx.x * 256 + threadIdx.x;
  const float* W; uint32_t* Wp; int nKt, Kdim, idx;
  if (tid < MT_ * KT1_ * 256) { W = W1; Wp = Wp1; nKt = KT1_; Kdim = K1_; idx = tid; }
  else {
    idx = tid - MT_ * KT1_ * 256;
    if (idx >= MT_ * KT2_ * 256) return;
    W = W2; Wp = Wp2; nKt = KT2_; Kdim = CO_;
  }
  int s = idx & 255, lane = s >> 3, v = s & 7;
  int kt = (idx >> 8) % nKt, mt = (idx >> 8) / nKt;
  int M  = mt * 16 + (lane & 15);
  int K0 = kt * 32 + ((lane >> 4) << 4) + 2 * v;
  Wp[idx] = pack2(W[(size_t)M * Kdim + K0], W[(size_t)M * Kdim + K0 + 1]);
}

// ---------------- 2) 3-NN + reciprocal-sum weights -------------------------
__global__ void __launch_bounds__(256)
fp_knn(const float* __restrict__ xyz_prev, const float* __restrict__ xyz_skip,
       float* __restrict__ w3, int* __restrict__ idx3) {
  __shared__ float sx[NP_], sy[NP_], sz[NP_];
  const int b = blockIdx.y;
  const float* xp = xyz_prev + (size_t)b * NP_ * 3;
  for (int j = threadIdx.x; j < NP_; j += 256) {
    sx[j] = xp[j * 3 + 0]; sy[j] = xp[j * 3 + 1]; sz[j] = xp[j * 3 + 2];
  }
  __syncthreads();
  const int n = blockIdx.x * 256 + threadIdx.x;
  const float* q = xyz_skip + ((size_t)b * NS_ + n) * 3;
  const float px = q[0], py = q[1], pz = q[2];
  float d0 = 3.4e38f, d1 = 3.4e38f, d2 = 3.4e38f;
  int   i0 = 0, i1 = 0, i2 = 0;
  float rsum = 0.0f;
  for (int j = 0; j < NP_; ++j) {
    float dx = px - sx[j], dy = py - sy[j], dz = pz - sz[j];
    float d  = sqrtf(dx * dx + dy * dy + dz * dz);
    rsum += 1.0f / (d + 1e-8f);
    if (d < d2) {
      if (d < d0)      { d2 = d1; i2 = i1; d1 = d0; i1 = i0; d0 = d; i0 = j; }
      else if (d < d1) { d2 = d1; i2 = i1; d1 = d;  i1 = j; }
      else             { d2 = d;  i2 = j; }
    }
  }
  const float inv = 1.0f / rsum;
  const size_t o = ((size_t)b * NS_ + n) * 3;
  w3[o + 0] = (1.0f / (d0 + 1e-8f)) * inv; idx3[o + 0] = i0;
  w3[o + 1] = (1.0f / (d1 + 1e-8f)) * inv; idx3[o + 1] = i1;
  w3[o + 2] = (1.0f / (d2 + 1e-8f)) * inv; idx3[o + 2] = i2;
}

// ---------------- 3) X1 = concat(interp, skip) as B-fragments (f16) --------
__global__ void __launch_bounds__(256)
fp_build_xpack1(const float* __restrict__ pp, const float* __restrict__ ps,
                const float* __restrict__ w3, const int* __restrict__ idx3,
                uint32_t* __restrict__ Xpack) {
  int tid = blockIdx.x * 256 + threadIdx.x;   // ((b*NT+nt)*KT1+kt)*256 + s
  int s = tid & 255, lane = s >> 3, v = s & 7;
  int t = tid >> 8;
  int kt = t % KT1_; t /= KT1_;
  int nt = t % NT_;  int b = t / NT_;
  int n  = nt * 16 + (lane & 15);
  int K0 = kt * 32 + ((lane >> 4) << 4) + 2 * v;
  float v0, v1;
  if (K0 < CP_) {
    size_t wo = ((size_t)b * NS_ + n) * 3;
    float wA = w3[wo], wB = w3[wo + 1], wC = w3[wo + 2];
    int   iA = idx3[wo], iB = idx3[wo + 1], iC = idx3[wo + 2];
    const float* base = pp + ((size_t)b * CP_ + K0) * NP_;
    v0 = wA * base[iA] + wB * base[iB] + wC * base[iC];
    base += NP_;
    v1 = wA * base[iA] + wB * base[iB] + wC * base[iC];
  } else {
    const float* base = ps + ((size_t)b * CS_ + (K0 - CP_)) * NS_ + n;
    v0 = base[0];
    v1 = base[NS_];
  }
  Xpack[tid] = pack2(v0, v1);
}

// ------- 4/7) fragment GEMM + fused BN partial sums ------------------------
__global__ void __launch_bounds__(256)
fp_gemm_frag(const uint32_t* __restrict__ Apack, const uint32_t* __restrict__ Bpack,
             float* __restrict__ Yfrag, float* __restrict__ partial, int nKt) {
  const int lane = threadIdx.x & 31;
  const int w    = threadIdx.x >> 5;        // 8 waves
  const int b    = blockIdx.y;
  const int ntB  = blockIdx.x * 8;          // 8 N-tiles per block
  const int mt0  = w * 2, mt1 = w * 2 + 1;  // 2 M-tiles per wave

  v8f zero = {};
  v8f acc[2][8];
#pragma unroll
  for (int i = 0; i < 2; ++i)
#pragma unroll
    for (int j = 0; j < 8; ++j) acc[i][j] = zero;

  for (int kt = 0; kt < nKt; ++kt) {
    const uint32_t* a0p = Apack + ((size_t)(mt0 * nKt + kt)) * 256 + lane * 8;
    const uint32_t* a1p = Apack + ((size_t)(mt1 * nKt + kt)) * 256 + lane * 8;
    v16h a0 = *(const v16h*)a0p;
    v16h a1 = *(const v16h*)a1p;
#pragma unroll
    for (int j = 0; j < 8; ++j) {
      const uint32_t* bp =
          Bpack + (((size_t)(b * NT_ + ntB + j)) * nKt + kt) * 256 + lane * 8;
      v16h bf = *(const v16h*)bp;
      acc[0][j] = __builtin_amdgcn_wmma_f32_16x16x32_f16(
          false, a0, false, bf, (short)0, acc[0][j], false, false);
      acc[1][j] = __builtin_amdgcn_wmma_f32_16x16x32_f16(
          false, a1, false, bf, (short)0, acc[1][j], false, false);
    }
  }
  // store D fragments
#pragma unroll
  for (int i = 0; i < 2; ++i) {
    int mt = w * 2 + i;
#pragma unroll
    for (int j = 0; j < 8; ++j) {
      float* yp = Yfrag + (((size_t)(b * MT_ + mt) * NT_ + ntB + j)) * 256 + lane * 8;
      *(v8f*)yp = acc[i][j];
    }
  }
  // fused BN partials: per-channel sum / sumsq over this block's 128 columns.
  // D layout: channel M = v + 8*(lane/16); xor shuffles <16 stay within a half.
  const int blockFlat = blockIdx.y * gridDim.x + blockIdx.x;
#pragma unroll
  for (int i = 0; i < 2; ++i) {
    float s[8], q[8];
#pragma unroll
    for (int v = 0; v < 8; ++v) {
      float t = 0.0f, tq = 0.0f;
#pragma unroll
      for (int j = 0; j < 8; ++j) { float x = acc[i][j][v]; t += x; tq += x * x; }
      s[v] = t; q[v] = tq;
    }
#pragma unroll
    for (int off = 1; off < 16; off <<= 1) {
#pragma unroll
      for (int v = 0; v < 8; ++v) {
        s[v] += __shfl_xor(s[v], off, 32);
        q[v] += __shfl_xor(q[v], off, 32);
      }
    }
    if ((lane & 15) == 0) {
      int mt  = w * 2 + i;
      int ch0 = mt * 16 + (lane >> 4) * 8;   // 8 channels per half-wave
      float* pb = partial + (size_t)blockFlat * (CO_ * 2) + ch0 * 2;
#pragma unroll
      for (int v = 0; v < 8; ++v) { pb[v * 2] = s[v]; pb[v * 2 + 1] = q[v]; }
    }
  }
}

// ---------------- 5/8) finalize BN stats from block partials ---------------
__global__ void __launch_bounds__(256)
fp_stats_finalize(const float* __restrict__ partial, float* __restrict__ stats) {
  int co = threadIdx.x;
  if (co >= CO_) return;
  float s = 0.0f, q = 0.0f;
  for (int p = 0; p < GBLK; ++p) {
    s += partial[(size_t)p * (CO_ * 2) + co * 2];
    q += partial[(size_t)p * (CO_ * 2) + co * 2 + 1];
  }
  const float inv = 1.0f / (float)(B_ * NS_);
  float mean = s * inv;
  float var  = q * inv - mean * mean;
  stats[co * 2 + 0] = mean;
  stats[co * 2 + 1] = rsqrtf(var + 1e-5f);
}

// ---------------- 6) BN+ReLU -> X2 B-fragments (f16) -----------------------
__global__ void __launch_bounds__(256)
fp_bn_relu_pack(const float* __restrict__ Yfrag, const float* __restrict__ stats,
                const float* __restrict__ gamma, const float* __restrict__ beta,
                uint32_t* __restrict__ Xpack2) {
  int tid = blockIdx.x * 256 + threadIdx.x;   // ((b*NT+nt)*KT2+kt)*256 + s
  int s = tid & 255, lane = s >> 3, v = s & 7;
  int t = tid >> 8;
  int kt = t % KT2_; t /= KT2_;
  int nt = t % NT_;  int b = t / NT_;
  int n16 = lane & 15;
  int co  = kt * 32 + ((lane >> 4) << 4) + 2 * v;
  float out[2];
#pragma unroll
  for (int u = 0; u < 2; ++u) {
    int c  = co + u;
    int mt = c >> 4, m = c & 15;
    const float* tile = Yfrag + (((size_t)(b * MT_ + mt) * NT_) + nt) * 256;
    float x = tile[(((m >> 3) << 4) + n16) * 8 + (m & 7)];
    float y = (x - stats[c * 2]) * stats[c * 2 + 1] * gamma[c] + beta[c];
    out[u] = fmaxf(y, 0.0f);
  }
  Xpack2[tid] = pack2(out[0], out[1]);
}

// ---------------- 9) BN+ReLU -> row-major f32 output -----------------------
__global__ void __launch_bounds__(256)
fp_bn_relu_out(const float* __restrict__ Yfrag, const float* __restrict__ stats,
               const float* __restrict__ gamma, const float* __restrict__ beta,
               float* __restrict__ outp) {
  int tid = blockIdx.x * 256 + threadIdx.x;   // (b*CO + co)*NS + n
  int n  = tid & (NS_ - 1);
  int co = (tid >> 12) & (CO_ - 1);
  int b  = tid >> 20;
  int mt = co >> 4, m = co & 15;
  int nt = n >> 4,  n16 = n & 15;
  const float* tile = Yfrag + (((size_t)(b * MT_ + mt) * NT_) + nt) * 256;
  float x = tile[(((m >> 3) << 4) + n16) * 8 + (m & 7)];
  float y = (x - stats[co * 2]) * stats[co * 2 + 1] * gamma[co] + beta[co];
  outp[tid] = fmaxf(y, 0.0f);
}

// ---------------------------------------------------------------------------
extern "C" void kernel_launch(void* const* d_in, const int* in_sizes, int n_in,
                              void* d_out, int out_size, void* d_ws, size_t ws_size,
                              hipStream_t stream) {
  (void)in_sizes; (void)n_in; (void)out_size; (void)ws_size;
  const float* xyz_prev    = (const float*)d_in[0];
  const float* xyz_skip    = (const float*)d_in[1];
  const float* points_prev = (const float*)d_in[2];
  const float* points_skip = (const float*)d_in[3];
  const float* W1 = (const float*)d_in[4];
  const float* g1 = (const float*)d_in[5];
  const float* b1 = (const float*)d_in[6];
  const float* W2 = (const float*)d_in[7];
  const float* g2 = (const float*)d_in[8];
  const float* b2 = (const float*)d_in[9];

  char* ws = (char*)d_ws;
  // workspace layout (bytes)
  const size_t OFF_W3   = 0;                                    // 786432
  const size_t OFF_I3   = OFF_W3   + (size_t)B_*NS_*3*4;        // 786432
  const size_t OFF_WP1  = OFF_I3   + (size_t)B_*NS_*3*4;        // 196608
  const size_t OFF_WP2  = OFF_WP1  + (size_t)MT_*KT1_*256*4;    // 131072
  const size_t OFF_XA   = OFF_WP2  + (size_t)MT_*KT2_*256*4;    // region A: X1 then X2 (50.3 MB)
  const size_t OFF_YB   = OFF_XA   + (size_t)B_*NT_*KT1_*256*4; // region B: Y1 then Y2 (67.1 MB)
  const size_t OFF_S1   = OFF_YB   + (size_t)B_*MT_*NT_*256*4;  // 2048
  const size_t OFF_S2   = OFF_S1   + (size_t)CO_*2*4;           // 2048
  const size_t OFF_PART = OFF_S2   + (size_t)CO_*2*4;           // 1 MB

  float*    w3    = (float*)(ws + OFF_W3);
  int*      idx3  = (int*)(ws + OFF_I3);
  uint32_t* Wp1   = (uint32_t*)(ws + OFF_WP1);
  uint32_t* Wp2   = (uint32_t*)(ws + OFF_WP2);
  uint32_t* XpA   = (uint32_t*)(ws + OFF_XA);
  float*    YfB   = (float*)(ws + OFF_YB);
  float*    st1   = (float*)(ws + OFF_S1);
  float*    st2   = (float*)(ws + OFF_S2);
  float*    part  = (float*)(ws + OFF_PART);

  // 1) pack weights
  {
    int total = (MT_ * KT1_ + MT_ * KT2_) * 256;  // 81920
    fp_pack_weights<<<dim3((total + 255) / 256), dim3(256), 0, stream>>>(W1, W2, Wp1, Wp2);
  }
  // 2) 3-NN weights
  fp_knn<<<dim3(NS_ / 256, B_), dim3(256), 0, stream>>>(xyz_prev, xyz_skip, w3, idx3);
  // 3) build X1 fragments
  fp_build_xpack1<<<dim3(B_ * NT_ * KT1_), dim3(256), 0, stream>>>(
      points_prev, points_skip, w3, idx3, XpA);
  // 4) GEMM1: y1 = W1 @ X1 (+ BN1 partials)
  fp_gemm_frag<<<dim3(NT_ / 8, B_), dim3(256), 0, stream>>>(Wp1, XpA, YfB, part, KT1_);
  // 5) BN1 stats finalize
  fp_stats_finalize<<<dim3(1), dim3(256), 0, stream>>>(part, st1);
  // 6) BN1+ReLU -> X2 fragments (region A reuse)
  fp_bn_relu_pack<<<dim3(B_ * NT_ * KT2_), dim3(256), 0, stream>>>(YfB, st1, g1, b1, XpA);
  // 7) GEMM2: y2 = W2 @ X2 (region B reuse, + BN2 partials)
  fp_gemm_frag<<<dim3(NT_ / 8, B_), dim3(256), 0, stream>>>(Wp2, XpA, YfB, part, KT2_);
  // 8) BN2 stats finalize
  fp_stats_finalize<<<dim3(1), dim3(256), 0, stream>>>(part, st2);
  // 9) BN2+ReLU -> d_out
  fp_bn_relu_out<<<dim3(B_ * CO_ * NS_ / 256), dim3(256), 0, stream>>>(
      YfB, st2, g2, b2, (float*)d_out);
}
